// _CrossAttentionFusion_57045755625519
// MI455X (gfx1250) — matrix operand order repaired and checked by reading, not compiled, not run on hardware
//
#include <hip/hip_runtime.h>
#include <hip/hip_bf16.h>

typedef _Float16 half_t;
typedef __attribute__((ext_vector_type(16))) _Float16 v16h;
typedef __attribute__((ext_vector_type(8)))  _Float16 v8h;
typedef __attribute__((ext_vector_type(8)))  float    v8f;
typedef __attribute__((ext_vector_type(4)))  float    v4f;

#define DEV __device__ __forceinline__

constexpr int B_  = 8;
constexpr int T_  = 2048;
constexpr int D_  = 128;
constexpr int H_  = 4;
constexpr int HD_ = 32;

union FragH { v16h v; v8h h[2]; };

DEV int lane_id() { return (int)(threadIdx.x & 31u); }

DEV v8f wmma_f16(v16h a, v16h b, v8f c) {
  // D = A(16x32 f16) * B(32x16 f16) + C(16x16 f32)
  return __builtin_amdgcn_wmma_f32_16x16x32_f16(false, a, false, b, (short)0, c,
                                                false, false);
}

// ---- CDNA5 async global->LDS copy (ASYNCcnt) ---------------------------
// Each active lane moves 16 bytes: LDS[lds_off] = MEM[gaddr].
DEV void async_copy_b128(unsigned int lds_off, const void* gaddr) {
  asm volatile("global_load_async_to_lds_b128 %0, %1, off"
               :: "v"(lds_off), "v"((unsigned long long)(uintptr_t)gaddr)
               : "memory");
}
DEV void wait_async0() {
  asm volatile("s_wait_asynccnt 0x0" ::: "memory");
}

// ---- Fragment loaders --------------------------------------------------
// A fragment (16x32, 16-bit): row-major source, row stride `stride` halves,
// K-window [koff, koff+32). Lane m (0-15): K koff+0..7 & koff+16..23;
// lanes 16-31: K koff+8..15 & koff+24..31.
DEV v16h load_a_f16p(const half_t* src, int stride, int koff) {
  int l = lane_id();
  int m = l & 15;
  int k0 = koff + ((l & 16) ? 8 : 0);
  const half_t* p = src + (size_t)m * stride + k0;
  FragH f;
  f.h[0] = *(const v8h*)(p);
  f.h[1] = *(const v8h*)(p + 16);
  return f.v;
}

DEV v16h load_a_f32p(const float* src, int stride, int koff) {
  int l = lane_id();
  int m = l & 15;
  int k0 = koff + ((l & 16) ? 8 : 0);
  const float* p = src + (size_t)m * stride + k0;
  FragH f;
#pragma unroll
  for (int i = 0; i < 8; ++i) {
    f.v[i]     = (half_t)p[i];
    f.v[8 + i] = (half_t)p[16 + i];
  }
  return f.v;
}

// B fragment (32K x 16N, 16-bit): source is "Bt" = N-major (row n of Bt is
// column n of B, contiguous in K), row stride `stride` halves.
// Lanes 0-15 (n): K koff+0..7 & koff+16..23; lanes 16-31 (n): K koff+8..15 & koff+24..31.
DEV v16h load_b_f16p(const half_t* bt, int stride, int koff) {
  int l = lane_id();
  int n = l & 15;
  int k0 = koff + ((l & 16) ? 8 : 0);
  const half_t* p = bt + (size_t)n * stride + k0;
  FragH f;
  f.h[0] = *(const v8h*)(p);
  f.h[1] = *(const v8h*)(p + 16);
  return f.v;
}

DEV v16h load_b_f32p(const float* bt, int stride, int koff) {
  int l = lane_id();
  int n = l & 15;
  int k0 = koff + ((l & 16) ? 8 : 0);
  const float* p = bt + (size_t)n * stride + k0;
  FragH f;
#pragma unroll
  for (int i = 0; i < 8; ++i) {
    f.v[i]     = (half_t)p[i];
    f.v[8 + i] = (half_t)p[16 + i];
  }
  return f.v;
}

// ========================================================================
// Kernel A: h = hawkes@Wp^T+bp ; q = lob@Winq^T (scaled) ; k = h@Wink^T ;
//           v = h@Winv^T.  Q,K stored (b,h,t,32) f16; V stored (b,h,32,t).
// lob tile staged into LDS with async copies overlapped with the h WMMA.
// ========================================================================
__global__ __launch_bounds__(256)
void caf_proj_qkv(const float* __restrict__ lob, const float* __restrict__ hawkes,
                  const float* __restrict__ Wp, const float* __restrict__ bp,
                  const float* __restrict__ Win, const float* __restrict__ bin,
                  half_t* __restrict__ Qg, half_t* __restrict__ Kg,
                  half_t* __restrict__ Vt) {
  __shared__ half_t hbuf[16][D_];
  __shared__ float  lobs[16][D_];

  int b   = blockIdx.x >> 7;
  int t0  = (blockIdx.x & 127) << 4;
  int tid = (int)threadIdx.x;
  int wave = tid >> 5;
  int l  = lane_id();
  int n  = l & 15;
  int m0 = (l & 16) ? 8 : 0;
  int col = wave * 16 + n;
  const float scale = 0.1767766952966369f;  // 1/sqrt(HD)

  const float* lobT = lob + ((size_t)b * T_ + t0) * D_;

  // ---- kick off async staging of the 16x128 lob tile (8 KB, 512 x 16B) ----
  {
    unsigned int lbase = (unsigned int)(uintptr_t)(&lobs[0][0]);
    const char*  gbase = (const char*)lobT;
#pragma unroll
    for (int c2 = tid; c2 < 512; c2 += 256)
      async_copy_b128(lbase + (unsigned int)c2 * 16u, gbase + (size_t)c2 * 16);
  }

  // ---- h projection (K = 32), overlapped with the async copy ----
  {
    v16h a  = load_a_f32p(hawkes + ((size_t)b * T_ + t0) * 32, 32, 0);
    v16h bb = load_b_f32p(Wp + (size_t)(wave * 16) * 32, 32, 0);
    v8f  c  = {};
    c = wmma_f16(a, bb, c);
    float bias = bp[col];
#pragma unroll
    for (int i = 0; i < 8; ++i) hbuf[m0 + i][col] = (half_t)(c[i] + bias);
  }
  wait_async0();
  __syncthreads();  // hbuf + lobs both visible

  // ---- q = lob @ Win[0:128]^T, pre-scaled ----
  {
    v8f c = {};
#pragma unroll
    for (int g = 0; g < 4; ++g) {
      v16h a  = load_a_f32p(&lobs[0][0], D_, 32 * g);
      v16h bb = load_b_f32p(Win + (size_t)(wave * 16) * D_, D_, 32 * g);
      c = wmma_f16(a, bb, c);
    }
    float bias = bin[col];
    int head = col >> 5, f = col & 31;
    half_t* qb = Qg + (((size_t)b * H_ + head) * T_ + t0) * HD_ + f;
#pragma unroll
    for (int i = 0; i < 8; ++i)
      qb[(size_t)(m0 + i) * HD_] = (half_t)((c[i] + bias) * scale);
  }

  // ---- k = h @ Win[128:256]^T ----
  {
    v8f c = {};
#pragma unroll
    for (int g = 0; g < 4; ++g) {
      v16h a  = load_a_f16p(&hbuf[0][0], D_, 32 * g);
      v16h bb = load_b_f32p(Win + (size_t)(D_ + wave * 16) * D_, D_, 32 * g);
      c = wmma_f16(a, bb, c);
    }
    float bias = bin[D_ + col];
    int head = col >> 5, f = col & 31;
    half_t* kb = Kg + (((size_t)b * H_ + head) * T_ + t0) * HD_ + f;
#pragma unroll
    for (int i = 0; i < 8; ++i)
      kb[(size_t)(m0 + i) * HD_] = (half_t)(c[i] + bias);
  }

  // ---- v = h @ Win[256:384]^T, stored transposed (b,h,32,t) ----
  {
    v8f c = {};
#pragma unroll
    for (int g = 0; g < 4; ++g) {
      v16h a  = load_a_f16p(&hbuf[0][0], D_, 32 * g);
      v16h bb = load_b_f32p(Win + (size_t)(2 * D_ + wave * 16) * D_, D_, 32 * g);
      c = wmma_f16(a, bb, c);
    }
    float bias = bin[2 * D_ + col];
    int head = col >> 5, f = col & 31;
    half_t* vb = Vt + (((size_t)b * H_ + head) * HD_ + f) * T_ + t0 + m0;
    v8h vv;
#pragma unroll
    for (int i = 0; i < 8; ++i) vv[i] = (half_t)(c[i] + bias);
    *(v8h*)vb = vv;  // rows of C are contiguous tokens here -> 16B store
  }
}

// ========================================================================
// Kernel B: per (b, 16-query tile): loop 4 heads; scores -> LDS, softmax,
// head-averaged attention accumulated in LDS f32, ctx via WMMA over LDS P.
// attn_w streamed to HBM exactly once, with non-temporal stores so the
// L2-resident K/V (8 MB) and ctx working set are not evicted.
// ========================================================================
__global__ __launch_bounds__(256)
void caf_attention(const half_t* __restrict__ Qg, const half_t* __restrict__ Kg,
                   const half_t* __restrict__ Vt, float* __restrict__ ctxg,
                   float* __restrict__ attnw) {
  __shared__ half_t e_buf[16][T_];      //  64 KB: scores, then exp(s - m)
  __shared__ float  avg[16][T_];        // 128 KB: head-averaged attention
  __shared__ float  ctxp[8][16][HD_];   //  16 KB: per-wave ctx partials
  __shared__ float  red[16][16];
  __shared__ float  row_inv[16];

  int b   = blockIdx.x >> 7;
  int t0  = (blockIdx.x & 127) << 4;
  int tid = (int)threadIdx.x;
  int wave = tid >> 5;
  int l  = lane_id();
  int n  = l & 15;
  int m0 = (l & 16) ? 8 : 0;
  int row = tid >> 4, sub = tid & 15;
  int cbase = sub * 128;

  float* avg_flat = &avg[0][0];
  for (int idx = tid; idx < 16 * T_; idx += 256) avg_flat[idx] = 0.0f;
  __syncthreads();

  for (int head = 0; head < H_; ++head) {
    size_t bh = (size_t)b * H_ + head;

    // ---- scores: S = Qtile(16x32) @ K^T, 128 key tiles of 16 ----
    v16h aq = load_a_f16p(Qg + (bh * T_ + t0) * HD_, HD_, 0);
    for (int kt = wave; kt < 128; kt += 8) {
      const half_t* kb = Kg + (bh * T_ + (size_t)kt * 16) * HD_;
      if (kt + 8 < 128)
        __builtin_prefetch(Kg + (bh * T_ + (size_t)(kt + 8) * 16) * HD_, 0, 1);
      v16h bk = load_b_f16p(kb, HD_, 0);
      v8f c = {};
      c = wmma_f16(aq, bk, c);
#pragma unroll
      for (int i = 0; i < 8; ++i) e_buf[m0 + i][kt * 16 + n] = (half_t)c[i];
    }
    __syncthreads();

    // ---- softmax: row max (vectorized ds_load_b128) ----
    float mx = -1e30f;
    for (int j = 0; j < 16; ++j) {
      v8h v = *(const v8h*)&e_buf[row][cbase + 8 * j];
#pragma unroll
      for (int i = 0; i < 8; ++i) mx = fmaxf(mx, (float)v[i]);
    }
    red[row][sub] = mx;
    __syncthreads();
    float rmx = -1e30f;
#pragma unroll
    for (int s2 = 0; s2 < 16; ++s2) rmx = fmaxf(rmx, red[row][s2]);
    __syncthreads();

    // ---- exp + sum (rewrite e_buf in place, vectorized) ----
    float ssum = 0.0f;
    for (int j = 0; j < 16; ++j) {
      v8h v = *(const v8h*)&e_buf[row][cbase + 8 * j];
      v8h o;
#pragma unroll
      for (int i = 0; i < 8; ++i) {
        float e = __expf((float)v[i] - rmx);
        o[i] = (half_t)e;
        ssum += e;
      }
      *(v8h*)&e_buf[row][cbase + 8 * j] = o;
    }
    red[row][sub] = ssum;
    __syncthreads();
    float tot = 0.0f;
#pragma unroll
    for (int s2 = 0; s2 < 16; ++s2) tot += red[row][s2];
    float inv = 1.0f / tot;
    if (sub == 0) row_inv[row] = inv;
    __syncthreads();

    // ---- accumulate head-averaged attention (vectorized) ----
    float qf = 0.25f * inv;
    for (int j = 0; j < 16; ++j) {
      v8h v = *(const v8h*)&e_buf[row][cbase + 8 * j];
      v4f* ap = (v4f*)&avg[row][cbase + 8 * j];
      v4f a0 = ap[0], a1 = ap[1];
#pragma unroll
      for (int i = 0; i < 4; ++i) a0[i] += (float)v[i] * qf;
#pragma unroll
      for (int i = 0; i < 4; ++i) a1[i] += (float)v[4 + i] * qf;
      ap[0] = a0; ap[1] = a1;
    }

    // ---- ctx = P @ V; wave w owns keys [w*256, w*256+256) ----
    v8f c0 = {}, c1 = {};
#pragma unroll
    for (int ch = 0; ch < 8; ++ch) {
      int kb2 = wave * 256 + ch * 32;
      v16h ap = load_a_f16p(&e_buf[0][0], T_, kb2);
      v16h b0 = load_b_f16p(Vt + (bh * HD_ + 0)  * T_, T_, kb2);
      v16h b1 = load_b_f16p(Vt + (bh * HD_ + 16) * T_, T_, kb2);
      c0 = wmma_f16(ap, b0, c0);
      c1 = wmma_f16(ap, b1, c1);
    }
#pragma unroll
    for (int i = 0; i < 8; ++i) {
      float iv = row_inv[m0 + i];
      ctxp[wave][m0 + i][n]      = c0[i] * iv;
      ctxp[wave][m0 + i][16 + n] = c1[i] * iv;
    }
    __syncthreads();

    for (int e2 = tid; e2 < 512; e2 += 256) {
      int r = e2 >> 5, cc = e2 & 31;
      float s = 0.0f;
#pragma unroll
      for (int w2 = 0; w2 < 8; ++w2) s += ctxp[w2][r][cc];
      ctxg[((size_t)b * T_ + t0 + r) * D_ + head * HD_ + cc] = s;
    }
    __syncthreads();
  }

  // ---- stream averaged attention tile to HBM once (non-temporal) ----
  v4f*       dst = (v4f*)(attnw + ((size_t)b * T_ + t0 + row) * T_ + cbase);
  const v4f* sp  = (const v4f*)(&avg[row][cbase]);
#pragma unroll
  for (int j2 = 0; j2 < 32; ++j2) __builtin_nontemporal_store(sp[j2], &dst[j2]);
}

// ========================================================================
// Kernel C: out = LayerNorm(lob + ctx @ Wo^T + bo) * gamma + beta
// ctx tile staged into LDS once via async copies (kills 8x redundant reads)
// ========================================================================
__global__ __launch_bounds__(256)
void caf_out_ln(const float* __restrict__ ctxg, const float* __restrict__ Wo,
                const float* __restrict__ bo, const float* __restrict__ lob,
                const float* __restrict__ gamma, const float* __restrict__ beta,
                float* __restrict__ outp) {
  __shared__ float ctbuf[16][D_];
  __shared__ float xbuf[16][D_];
  __shared__ float red2[16][16];

  int b   = blockIdx.x >> 7;
  int t0  = (blockIdx.x & 127) << 4;
  int tid = (int)threadIdx.x;
  int wave = tid >> 5;
  int l  = lane_id();
  int n  = l & 15;
  int m0 = (l & 16) ? 8 : 0;
  int col = wave * 16 + n;

  // ---- async-stage the 16x128 ctx tile (8 KB, 512 x 16B) ----
  {
    unsigned int lbase = (unsigned int)(uintptr_t)(&ctbuf[0][0]);
    const char*  gbase = (const char*)(ctxg + ((size_t)b * T_ + t0) * D_);
#pragma unroll
    for (int c2 = tid; c2 < 512; c2 += 256)
      async_copy_b128(lbase + (unsigned int)c2 * 16u, gbase + (size_t)c2 * 16);
  }
  wait_async0();
  __syncthreads();

  v8f c = {};
#pragma unroll
  for (int g = 0; g < 4; ++g) {
    v16h a  = load_a_f32p(&ctbuf[0][0], D_, 32 * g);
    v16h bb = load_b_f32p(Wo + (size_t)(wave * 16) * D_, D_, 32 * g);
    c = wmma_f16(a, bb, c);
  }
  float bias = bo[col];
#pragma unroll
  for (int i = 0; i < 8; ++i) xbuf[m0 + i][col] = c[i] + bias;
  __syncthreads();

  int row = tid >> 4, sub = tid & 15;
  const float* lrow = lob + ((size_t)b * T_ + t0 + row) * D_;
  float x[8];
  float s = 0.0f, s2 = 0.0f;
#pragma unroll
  for (int j = 0; j < 8; ++j) {
    int cc = sub * 8 + j;
    float v = xbuf[row][cc] + lrow[cc];
    x[j] = v; s += v; s2 += v * v;
  }
  red2[row][sub] = s;
  __syncthreads();
  float mean = 0.0f;
#pragma unroll
  for (int t = 0; t < 16; ++t) mean += red2[row][t];
  mean *= (1.0f / 128.0f);
  __syncthreads();
  red2[row][sub] = s2;
  __syncthreads();
  float m2 = 0.0f;
#pragma unroll
  for (int t = 0; t < 16; ++t) m2 += red2[row][t];
  m2 *= (1.0f / 128.0f);
  float rstd = rsqrtf(m2 - mean * mean + 1e-5f);

  float* orow = outp + ((size_t)b * T_ + t0 + row) * D_;
#pragma unroll
  for (int j = 0; j < 8; ++j) {
    int cc = sub * 8 + j;
    orow[cc] = (x[j] - mean) * rstd * gamma[cc] + beta[cc];
  }
}

// ========================================================================
extern "C" void kernel_launch(void* const* d_in, const int* in_sizes, int n_in,
                              void* d_out, int out_size, void* d_ws, size_t ws_size,
                              hipStream_t stream) {
  const float* lob    = (const float*)d_in[0];
  const float* hawkes = (const float*)d_in[1];
  const float* Wp     = (const float*)d_in[2];
  const float* bp     = (const float*)d_in[3];
  const float* Win    = (const float*)d_in[4];
  const float* bin    = (const float*)d_in[5];
  const float* Wo     = (const float*)d_in[6];
  const float* bo     = (const float*)d_in[7];
  const float* gamma  = (const float*)d_in[8];
  const float* beta   = (const float*)d_in[9];

  constexpr size_t QKV = (size_t)B_ * H_ * T_ * HD_;  // 2,097,152 halves each
  half_t* Qg   = (half_t*)d_ws;
  half_t* Kg   = Qg + QKV;
  half_t* Vt   = Kg + QKV;
  float*  ctxg = (float*)(Vt + QKV);                  // 8 MB f32

  float* outp  = (float*)d_out;
  float* attnw = outp + (size_t)B_ * T_ * D_;

  dim3 grid(B_ * (T_ / 16));  // 1024 workgroups
  dim3 block(256);            // 8 wave32

  caf_proj_qkv<<<grid, block, 0, stream>>>(lob, hawkes, Wp, bp, Win, bin,
                                           Qg, Kg, Vt);
  caf_attention<<<grid, block, 0, stream>>>(Qg, Kg, Vt, ctxg, attnw);
  caf_out_ln<<<grid, block, 0, stream>>>(ctxg, Wo, bo, lob, gamma, beta, outp);
}
